// AdditiveAttention_87703232184988
// MI455X (gfx1250) — matrix operand order repaired and checked by reading, not compiled, not run on hardware
//
#include <hip/hip_runtime.h>
#include <hip/hip_bf16.h>

// ---------------------------------------------------------------------------
// Additive (Bahdanau) attention, B=32, T=4096, H=512, fused for MI455X.
// Main GEMM (enc_seq @ W_h^T, 68.7 GFLOP) runs on v_wmma_f32_16x16x32_bf16
// with f32 accumulation; tanh/dot-v epilogue fused into the WMMA tile loop.
// W_h slab staging uses GLOBAL_LOAD_ASYNC_TO_LDS (ASYNCcnt-tracked) — the
// builtin exists on this toolchain; expected params are v4i* in AS1/AS3.
// ---------------------------------------------------------------------------

typedef __attribute__((ext_vector_type(16))) __bf16 v16bf;
typedef __attribute__((ext_vector_type(8)))  float  v8f;
typedef __attribute__((ext_vector_type(4)))  int    v4i;

#define HIDN  512
#define TLEN  4096
#define BATCH 32

#if defined(__gfx1250__) && \
    __has_builtin(__builtin_amdgcn_global_load_async_to_lds_b128) && \
    __has_builtin(__builtin_amdgcn_s_wait_asynccnt)
#define USE_ASYNC_LDS 1
#else
#define USE_ASYNC_LDS 0
#endif

// ---- W_h f32 -> bf16 (one-time per call; stays resident in L2) -------------
__global__ __launch_bounds__(256) void convert_wh(const float* __restrict__ W,
                                                  __bf16* __restrict__ Wb) {
  int i = blockIdx.x * 256 + threadIdx.x;        // grid covers HIDN*HIDN
  Wb[i] = (__bf16)W[i];
}

// ---- s[b,k] = dot(dec_state[b,:], W_s[k,:])  (tiny: 16.8 MFLOP) -------------
__global__ __launch_bounds__(256) void dec_proj(const float* __restrict__ dec,
                                                const float* __restrict__ Ws,
                                                float* __restrict__ s) {
  int b = blockIdx.x;
  const float* d = dec + (size_t)b * HIDN;
  #pragma unroll
  for (int kk = 0; kk < 2; ++kk) {
    int k = threadIdx.x + kk * 256;
    const float* w = Ws + (size_t)k * HIDN;
    float acc = 0.f;
    for (int h = 0; h < HIDN; h += 4) {
      float4 wv = *(const float4*)(w + h);
      float4 dv = *(const float4*)(d + h);
      acc += wv.x * dv.x + wv.y * dv.y + wv.z * dv.z + wv.w * dv.w;
    }
    s[b * HIDN + k] = acc;
  }
}

// ---- main: e[b,t] = sum_n v[n] * tanh( (enc @ W_h^T)[b,t,n] + s[b,n] ) -----
// Block: 256 threads = 8 waves; one b, 128 T-rows per block, 16 rows/wave.
// A panel (16 x 512) held in registers as bf16; W_h N-slabs double-buffered
// through LDS and shared by all 8 waves.
__global__ __launch_bounds__(256) void attn_score_wmma(
    const float*  __restrict__ enc,   // [B,T,H] f32
    const __bf16* __restrict__ Whb,   // [H,H]   bf16 (row n, col k)
    const float*  __restrict__ s,     // [B,H]
    const float*  __restrict__ v,     // [H]
    float*        __restrict__ e)     // [B,T]   (written into d_out 'a' slot)
{
  __shared__ __align__(32) __bf16 slab[2][16 * HIDN];   // 2 x 16KB

  const int b    = blockIdx.x >> 5;            // 32 blocks per batch element
  const int t0b  = (blockIdx.x & 31) * 128;
  const int wave = threadIdx.x >> 5;
  const int lane = threadIdx.x & 31;
  const int t0   = t0b + wave * 16;            // this wave's 16 rows
  const int half = lane >> 4;                  // 0 | 1
  const int l15  = lane & 15;

  // --- load the 16x512 A panel once, converting f32 -> bf16 in registers.
  // A k order used here (contiguous, same as B; 16x32 per k-step):
  //   lanes 0-15  : row M=lane,     elems 0..15 = K(ks*32+0..15)
  //   lanes 16-31 : row M=lane-16,  elems 0..15 = K(ks*32+16..31)
  v16bf A[16];
  {
    const float* rowp = enc + (((size_t)b * TLEN) + (size_t)(t0 + l15)) * HIDN;
    #pragma unroll
    for (int ks = 0; ks < 16; ++ks) {
      const float* p = rowp + ks * 32 + half * 16;
      v16bf aa;
      #pragma unroll
      for (int j = 0; j < 8; ++j) aa[j]     = (__bf16)p[j];
      #pragma unroll
      for (int j = 0; j < 8; ++j) aa[8 + j] = (__bf16)p[8 + j];
      A[ks] = aa;
    }
  }

  // --- cooperative stage of one W_h N-slab (rows n*16..n*16+15, all K) ------
  // Each of the 256 threads moves 64B; async path: 4x async-b128 to LDS
  // (inst offset applies to BOTH global and LDS addresses per ISA §10).
  auto stage = [&](int n, int buf) {
    int row = threadIdx.x >> 4;                 // 0..15
    int seg = threadIdx.x & 15;                 // 32 bf16 (64B) per thread
    const __bf16* src = Whb + ((size_t)(n * 16 + row)) * HIDN + seg * 32;
    __bf16*       dst = &slab[buf][row * HIDN + seg * 32];
#if USE_ASYNC_LDS
    auto gsrc = (__attribute__((address_space(1))) v4i*)(src);
    auto ldst = (__attribute__((address_space(3))) v4i*)(dst);
    __builtin_amdgcn_global_load_async_to_lds_b128(gsrc, ldst, 0,  0);
    __builtin_amdgcn_global_load_async_to_lds_b128(gsrc, ldst, 16, 0);
    __builtin_amdgcn_global_load_async_to_lds_b128(gsrc, ldst, 32, 0);
    __builtin_amdgcn_global_load_async_to_lds_b128(gsrc, ldst, 48, 0);
#else
    const int4* s4 = (const int4*)src;
    int4*       d4 = (int4*)dst;
    d4[0] = s4[0]; d4[1] = s4[1]; d4[2] = s4[2]; d4[3] = s4[3];
#endif
  };

  float ep[8] = {0.f, 0.f, 0.f, 0.f, 0.f, 0.f, 0.f, 0.f};

  stage(0, 0);
  for (int n = 0; n < 32; ++n) {
#if USE_ASYNC_LDS
    __builtin_amdgcn_s_wait_asynccnt(0);        // own async->LDS writes done
#endif
    __syncthreads();                            // slab[n&1] ready; prior reads done
    if (n + 1 < 32) stage(n + 1, (n + 1) & 1);

    // B fragment: lane holds column N=l15 (slab row l15), K-half by `half`.
    const __bf16* base = &slab[n & 1][l15 * HIDN + half * 16];

    v8f acc = {};
    #pragma unroll
    for (int ks = 0; ks < 16; ++ks) {
      v16bf bfrag = *(const v16bf*)(base + ks * 32);
      acc = __builtin_amdgcn_wmma_f32_16x16x32_bf16(
          false, A[ks], false, bfrag, (short)0, acc, false, false);
    }

    // epilogue: C/D layout — lane l: col N = l&15, VGPR r: row r + 8*(l>=16)
    int   col = n * 16 + l15;
    float sv  = s[b * HIDN + col];
    float vv  = v[col];
    #pragma unroll
    for (int r = 0; r < 8; ++r)
      ep[r] += vv * tanhf(acc[r] + sv);
  }

  // reduce across the 16 lanes holding the same rows (xor masks stay in-half)
  #pragma unroll
  for (int r = 0; r < 8; ++r) {
    float x = ep[r];
    x += __shfl_xor(x, 1, 32);
    x += __shfl_xor(x, 2, 32);
    x += __shfl_xor(x, 4, 32);
    x += __shfl_xor(x, 8, 32);
    ep[r] = x;
  }
  if (l15 == 0) {
    #pragma unroll
    for (int r = 0; r < 8; ++r)
      e[(size_t)b * TLEN + t0 + half * 8 + r] = ep[r];
  }
}

// ---- masked softmax over T, in place ---------------------------------------
__global__ __launch_bounds__(256) void softmax_mask(float* __restrict__ a,
                                                    const int* __restrict__ mask) {
  int b = blockIdx.x;
  __shared__ float redm[8];
  __shared__ float reds[8];
  float loc[16];
  float mx = -3.0e38f;
  #pragma unroll
  for (int i = 0; i < 16; ++i) {
    int t = threadIdx.x + i * 256;
    float ev = a[(size_t)b * TLEN + t];
    ev = (mask[b * TLEN + t] == 0) ? -1.0e9f : ev;
    loc[i] = ev;
    mx = fmaxf(mx, ev);
  }
  #pragma unroll
  for (int m = 16; m >= 1; m >>= 1) mx = fmaxf(mx, __shfl_xor(mx, m, 32));
  if ((threadIdx.x & 31) == 0) redm[threadIdx.x >> 5] = mx;
  __syncthreads();
  float gmx = redm[0];
  #pragma unroll
  for (int i = 1; i < 8; ++i) gmx = fmaxf(gmx, redm[i]);

  float sum = 0.f;
  #pragma unroll
  for (int i = 0; i < 16; ++i) {
    loc[i] = __expf(loc[i] - gmx);
    sum += loc[i];
  }
  #pragma unroll
  for (int m = 16; m >= 1; m >>= 1) sum += __shfl_xor(sum, m, 32);
  if ((threadIdx.x & 31) == 0) reds[threadIdx.x >> 5] = sum;
  __syncthreads();
  float tot = 0.f;
  #pragma unroll
  for (int i = 0; i < 8; ++i) tot += reds[i];
  float inv = 1.0f / tot;
  #pragma unroll
  for (int i = 0; i < 16; ++i)
    a[(size_t)b * TLEN + threadIdx.x + i * 256] = loc[i] * inv;
}

// ---- ctx = a @ enc_seq  (second HBM streaming pass) ------------------------
__global__ __launch_bounds__(256) void ctx_zero(float* __restrict__ ctx) {
  int i = blockIdx.x * 256 + threadIdx.x;
  if (i < BATCH * HIDN) ctx[i] = 0.f;
}

__global__ __launch_bounds__(256) void ctx_accum(const float* __restrict__ enc,
                                                 const float* __restrict__ a,
                                                 float* __restrict__ ctx) {
  int b      = blockIdx.x >> 4;                // 16 T-chunks of 256 per b
  int tchunk = blockIdx.x & 15;
  int h      = threadIdx.x;                    // 0..255; owns h and h+256
  const float* encb = enc + ((size_t)b * TLEN + (size_t)tchunk * 256) * HIDN;
  const float* ab   = a + (size_t)b * TLEN + tchunk * 256;
  float acc0 = 0.f, acc1 = 0.f;
  for (int t = 0; t < 256; ++t) {
    float at = ab[t];
    acc0 += at * encb[(size_t)t * HIDN + h];
    acc1 += at * encb[(size_t)t * HIDN + h + 256];
  }
  unsafeAtomicAdd(&ctx[b * HIDN + h],       acc0);
  unsafeAtomicAdd(&ctx[b * HIDN + h + 256], acc1);
}

// ---------------------------------------------------------------------------
extern "C" void kernel_launch(void* const* d_in, const int* in_sizes, int n_in,
                              void* d_out, int out_size, void* d_ws, size_t ws_size,
                              hipStream_t stream) {
  const float* enc  = (const float*)d_in[0];   // [B,T,H]
  const int*   mask = (const int*)  d_in[1];   // [B,T]
  const float* dec  = (const float*)d_in[2];   // [B,H]
  const float* W_h  = (const float*)d_in[3];   // [H,H]
  const float* W_s  = (const float*)d_in[4];   // [H,H]
  const float* vv   = (const float*)d_in[5];   // [H]

  float* out  = (float*)d_out;
  float* ctx  = out;                           // [B,H]
  float* aArr = out + BATCH * HIDN;            // [B,T]

  // workspace layout: [W_h bf16: 512KB][s f32: 64KB]
  __bf16* Whb = (__bf16*)d_ws;
  float*  sB  = (float*)((char*)d_ws + (size_t)HIDN * HIDN * sizeof(__bf16));

  convert_wh<<<(HIDN * HIDN) / 256, 256, 0, stream>>>(W_h, Whb);
  dec_proj<<<BATCH, 256, 0, stream>>>(dec, W_s, sB);

  // scores -> aArr (as raw e values)
  attn_score_wmma<<<BATCH * (TLEN / 128), 256, 0, stream>>>(enc, Whb, sB, vv, aArr);

  softmax_mask<<<BATCH, 256, 0, stream>>>(aArr, mask);

  ctx_zero<<<(BATCH * HIDN + 255) / 256, 256, 0, stream>>>(ctx);
  ctx_accum<<<BATCH * 16, 256, 0, stream>>>(enc, aArr, ctx);
}
